// CausalSelfAttention_38457137168858
// MI455X (gfx1250) — matrix operand order repaired and checked by reading, not compiled, not run on hardware
//
#include <hip/hip_runtime.h>
#include <hip/hip_bf16.h>

// ---------------------------------------------------------------------------
// GPT-2 causal self-attention for MI455X (gfx1250, wave32, WMMA + TDM).
// Matmuls: v_wmma_f32_16x16x32_f16 (f16 in, f32 accumulate).
// Global->LDS tile staging: tensor_load_to_lds (TDM), double-buffered in the
// GEMMs so the DMA of stage i+1 overlaps the WMMAs of stage i.
//
// WMMA register layouts assumed (CDNA5 ISA 7.12.2):
//   A (16x32 f16): lane L -> M = L&15 ; elem i -> K = ks*32 + (i>=8)*16 + (L>>4)*8 + (i&7)
//   B (32x16 f16): lane L -> N = L&15 ; elem i -> K = ks*32 + (i>=8)*16 + (L>>4)*8 + (i&7)
//   C/D (16x16 f32): lane L -> N = L&15 ; vgpr r -> M = r + 8*(L>>4)
// ---------------------------------------------------------------------------

typedef __attribute__((ext_vector_type(16))) _Float16 v16h;
typedef __attribute__((ext_vector_type(8)))  _Float16 v8h;
typedef __attribute__((ext_vector_type(4)))  _Float16 v4h;
typedef __attribute__((ext_vector_type(8)))  float    v8f;
typedef __attribute__((ext_vector_type(4)))  unsigned int v4u;
typedef __attribute__((ext_vector_type(8)))  int      v8i;
typedef __attribute__((ext_vector_type(4)))  int      v4i;

#define NB  4
#define NS  2048
#define ND  1024
#define NH  16
#define NHD 64
#define NTOK (NB * NS)          // 8192 rows

__device__ __forceinline__ v16h cat16(v8h lo, v8h hp) {
  return __builtin_shufflevector(lo, hp, 0,1,2,3,4,5,6,7,8,9,10,11,12,13,14,15);
}
__device__ __forceinline__ v8f wmma_f16(v16h a, v16h b, v8f c) {
  return __builtin_amdgcn_wmma_f32_16x16x32_f16(false, a, false, b, (short)0, c,
                                                false, false);
}
__device__ __forceinline__ float rmax16(float v) {
  v = fmaxf(v, __shfl_xor(v, 1, 32));
  v = fmaxf(v, __shfl_xor(v, 2, 32));
  v = fmaxf(v, __shfl_xor(v, 4, 32));
  v = fmaxf(v, __shfl_xor(v, 8, 32));
  return v;
}
__device__ __forceinline__ float rsum16(float v) {
  v += __shfl_xor(v, 1, 32);
  v += __shfl_xor(v, 2, 32);
  v += __shfl_xor(v, 4, 32);
  v += __shfl_xor(v, 8, 32);
  return v;
}
// Low 32 bits of a generic pointer to __shared__ == LDS byte offset
__device__ __forceinline__ unsigned int lds_off(const void* p) {
  return (unsigned int)(uintptr_t)p;
}

// ------------------- Tensor Data Mover: 2D f16 tile load -------------------
// Loads tile [tileD1 rows][tileD0 halves] from row-major gptr (row stride
// strideElems halves) into LDS at ldsOff, padding each 64-half row by 16
// halves (pad_interval = 32 DWORDs, pad_amount = 8 DWORDs) -> LDS stride 80.
__device__ __forceinline__ void tdm_load_2d(unsigned int ldsOff,
                                            const _Float16* gptr,
                                            unsigned long long strideElems,
                                            unsigned int tileD0,
                                            unsigned int tileD1) {
  unsigned long long ga = (unsigned long long)(uintptr_t)gptr;
  v4u g0;
  g0[0] = 1u;                                        // count=1, user descriptor
  g0[1] = ldsOff;                                    // lds_addr
  g0[2] = (unsigned int)ga;                          // global_addr[31:0]
  g0[3] = (unsigned int)(ga >> 32) | (2u << 30);     // global_addr[56:32]|type=2
  const unsigned int TDBIG = 1u << 20;               // generous OOB dims
  v8i g1;
  g1[0] = (int)((1u << 16) |                         // data_size = 2B
                (1u << 20) |                         // pad_enable
                (4u << 22) |                         // pad_interval: 32 DWORDs
                (7u << 25));                         // pad_amount: 8 DWORDs
  g1[1] = (int)((TDBIG & 0xFFFFu) << 16);            // tensor_dim0[15:0]
  g1[2] = (int)((TDBIG >> 16) | ((TDBIG & 0xFFFFu) << 16)); // td0 hi | td1 lo
  g1[3] = (int)((TDBIG >> 16) | (tileD0 << 16));     // td1 hi | tile_dim0
  g1[4] = (int)tileD1;                               // tile_dim1 (tile_dim2=0)
  g1[5] = (int)(unsigned int)strideElems;            // dim0_stride[31:0]
  g1[6] = (int)(unsigned int)(strideElems >> 32);    // dim0_stride[47:32]
  g1[7] = 0;
  v4i z4 = {0, 0, 0, 0};
  v8i z8 = {0, 0, 0, 0, 0, 0, 0, 0};
  // 6-arg form (clang-23 / therock headers): (g0, g1, g2, g3, extra, cpol)
  __builtin_amdgcn_tensor_load_to_lds(g0, g1, z4, z4, z8, 0);
}

// ------------------------- fp32 -> fp16 convert ----------------------------
__global__ void cvt_f32_f16(const float* __restrict__ x, _Float16* __restrict__ y,
                            int n4) {
  int i = blockIdx.x * blockDim.x + threadIdx.x;
  if (i < n4) {
    float4 v = ((const float4*)x)[i];
    v4h o = {(_Float16)v.x, (_Float16)v.y, (_Float16)v.z, (_Float16)v.w};
    ((v4h*)y)[i] = o;
  }
}

// --------------- fp32 W[K][N] -> fp16 Wt[N][K] (transpose) -----------------
__global__ void xpose_f32_f16(const float* __restrict__ W, _Float16* __restrict__ Wt,
                              int K, int N) {
  __shared__ float tile[32][33];
  const int nBase = blockIdx.x * 32, kBase = blockIdx.y * 32;
  const int tx = threadIdx.x & 31, ty = threadIdx.x >> 5;   // 32 x 8
#pragma unroll
  for (int i = 0; i < 4; ++i) {
    int k = ty + i * 8;
    tile[k][tx] = W[(size_t)(kBase + k) * N + nBase + tx];
  }
  __syncthreads();
#pragma unroll
  for (int i = 0; i < 4; ++i) {
    int n = ty + i * 8;
    Wt[(size_t)(nBase + n) * K + kBase + tx] = (_Float16)tile[tx][n];
  }
}

// ------------------------------ WMMA GEMM ----------------------------------
// C[M][N] = A[M][K] @ Bt[N][K]^T + bias ; block 128x128, K-step 64.
// TDM stages both tiles into double-buffered LDS; wave 0 drives the DMA.
template <int OUT_HALF, int SCALE_Q>
__global__ __launch_bounds__(256) void gemm_wmma(
    const _Float16* __restrict__ A, const _Float16* __restrict__ Bt,
    const float* __restrict__ bias, void* __restrict__ Cv, int M, int N, int K) {
  __shared__ _Float16 As[2][128 * 80];   // stride 80 halves (160B, b128-aligned)
  __shared__ _Float16 Bs[2][128 * 80];
  const unsigned int STAGE = 128 * 80 * 2;   // bytes per buffer

  const int tid = threadIdx.x;
  const int lane = tid & 31, wid = tid >> 5;
  const int l16 = lane & 15, hi = lane >> 4;
  const int waveM = wid >> 1, waveN = wid & 1;    // 4x2 wave grid
  const int rowBase = blockIdx.y * 128;
  const int colBase = blockIdx.x * 128;
  const unsigned int asOff = lds_off(&As[0][0]);
  const unsigned int bsOff = lds_off(&Bs[0][0]);

  v8f acc[2][4];
#pragma unroll
  for (int mt = 0; mt < 2; ++mt)
#pragma unroll
    for (int nt = 0; nt < 4; ++nt)
#pragma unroll
      for (int e = 0; e < 8; ++e) acc[mt][nt][e] = 0.f;

  const int nIter = K >> 6;
  if (wid == 0) {    // prologue: stage 0
    tdm_load_2d(asOff, A + (size_t)rowBase * K, (unsigned long long)K, 64, 128);
    tdm_load_2d(bsOff, Bt + (size_t)colBase * K, (unsigned long long)K, 64, 128);
  }

  for (int it = 0; it < nIter; ++it) {
    const int buf = it & 1;
    if (wid == 0) __builtin_amdgcn_s_wait_tensorcnt(0);   // stage `it` landed
    __syncthreads();   // LDS visible to all waves; prev compute done everywhere
    if (wid == 0 && it + 1 < nIter) {   // DMA next stage while we do WMMAs
      int k0n = (it + 1) << 6;
      tdm_load_2d(asOff + (buf ^ 1) * STAGE, A + (size_t)rowBase * K + k0n,
                  (unsigned long long)K, 64, 128);
      tdm_load_2d(bsOff + (buf ^ 1) * STAGE, Bt + (size_t)colBase * K + k0n,
                  (unsigned long long)K, 64, 128);
    }
    const _Float16* as = As[buf];
    const _Float16* bs = Bs[buf];
#pragma unroll
    for (int ks = 0; ks < 2; ++ks) {
      v16h af[2], bf[4];
#pragma unroll
      for (int mt = 0; mt < 2; ++mt) {
        const _Float16* p = as + (waveM * 32 + mt * 16 + l16) * 80 + ks * 32;
        af[mt] = cat16(*(const v8h*)(p + hi * 8), *(const v8h*)(p + 16 + hi * 8));
      }
#pragma unroll
      for (int nt = 0; nt < 4; ++nt) {
        const _Float16* p = bs + (waveN * 64 + nt * 16 + l16) * 80 + ks * 32 + hi * 8;
        bf[nt] = cat16(*(const v8h*)p, *(const v8h*)(p + 16));
      }
#pragma unroll
      for (int mt = 0; mt < 2; ++mt)
#pragma unroll
        for (int nt = 0; nt < 4; ++nt)
          acc[mt][nt] = wmma_f16(af[mt], bf[nt], acc[mt][nt]);
    }
  }

#pragma unroll
  for (int nt = 0; nt < 4; ++nt) {
    int n = colBase + waveN * 64 + nt * 16 + l16;
    float bv = bias[n];
    float sc = (SCALE_Q && n < ND) ? 0.125f : 1.0f;   // 1/sqrt(64)
#pragma unroll
    for (int mt = 0; mt < 2; ++mt)
#pragma unroll
      for (int r = 0; r < 8; ++r) {
        int m = rowBase + waveM * 32 + mt * 16 + r + 8 * hi;
        float v = (acc[mt][nt][r] + bv) * sc;
        if (OUT_HALF)
          ((_Float16*)Cv)[(size_t)m * N + n] = (_Float16)v;
        else
          ((float*)Cv)[(size_t)m * N + n] = v;
      }
  }
}

// ------------------------- Flash attention (causal) ------------------------
// grid = (S/128, B*H); block = 256 (8 waves); each wave owns 16 query rows.
// QKV: f16 [NTOK][3*ND], Q pre-scaled by 1/sqrt(HD). O: f16 [NTOK][ND].
// K tile staged by TDM; V tile transposed manually (TDM has no transpose).
__global__ __launch_bounds__(256) void attn_wmma(const _Float16* __restrict__ QKV,
                                                 _Float16* __restrict__ O) {
  const int qb = blockIdx.x;
  const int bh = blockIdx.y;
  const int b = bh >> 4, h = bh & 15;
  const int tid = threadIdx.x;
  const int lane = tid & 31, wid = tid >> 5;
  const int l16 = lane & 15, hi = lane >> 4;

  __shared__ _Float16 Ksm[32 * 80];        // [key][hd]  stride 80 (TDM pad)
  __shared__ _Float16 Vts[64 * 40];        // [hd][key]  stride 40 (transposed)
  __shared__ _Float16 Psm[8 * 16 * 40];    // per-wave P tile [16][32] stride 40

  const unsigned int ksmOff = lds_off(&Ksm[0]);
  const int qRow0 = qb * 128 + wid * 16;
  const int qiMax = qRow0 + 15;

  // preload Q A-fragments (1/8 scale already folded in by the QKV GEMM)
  v16h qf[2];
  {
    const _Float16* qp = QKV + (size_t)(b * NS + qRow0 + l16) * (3 * ND) + h * NHD;
#pragma unroll
    for (int ks = 0; ks < 2; ++ks)
      qf[ks] = cat16(*(const v8h*)(qp + ks * 32 + hi * 8),
                     *(const v8h*)(qp + ks * 32 + 16 + hi * 8));
  }

  v8f o[4];
  float rowM[8], rowL[8];
#pragma unroll
  for (int nt = 0; nt < 4; ++nt)
#pragma unroll
    for (int e = 0; e < 8; ++e) o[nt][e] = 0.f;
#pragma unroll
  for (int r = 0; r < 8; ++r) { rowM[r] = -1e30f; rowL[r] = 0.f; }

  const _Float16* Kg = QKV + (size_t)(b * NS) * (3 * ND) + ND + h * NHD;
  const _Float16* Vg = QKV + (size_t)(b * NS) * (3 * ND) + 2 * ND + h * NHD;
  const int nkb = (qb + 1) * 4;            // 32-key blocks up to causal frontier

  for (int kb = 0; kb < nkb; ++kb) {
    // --- stage K via TDM (overlaps the manual V transpose below) ---
    if (wid == 0)
      tdm_load_2d(ksmOff, Kg + (size_t)(kb * 32) * (3 * ND),
                  (unsigned long long)(3 * ND), 64, 32);
    {   // V transposed into LDS: [hd][key]
      int key = tid >> 3, cch = (tid & 7) * 8;
      v8h vv = *(const v8h*)(Vg + (size_t)(kb * 32 + key) * (3 * ND) + cch);
#pragma unroll
      for (int j = 0; j < 8; ++j) Vts[(cch + j) * 40 + key] = vv[j];
    }
    if (wid == 0) __builtin_amdgcn_s_wait_tensorcnt(0);
    __syncthreads();

    if (kb * 32 <= qiMax) {                // skip fully-masked key blocks
      // scores S = Q @ K^T  (two 16-key tiles, two 32-deep k steps)
      v8f acc[2];
#pragma unroll
      for (int nt = 0; nt < 2; ++nt) {
#pragma unroll
        for (int e = 0; e < 8; ++e) acc[nt][e] = 0.f;
#pragma unroll
        for (int ks = 0; ks < 2; ++ks) {
          const _Float16* p = Ksm + (nt * 16 + l16) * 80 + ks * 32 + hi * 8;
          v16h kf = cat16(*(const v8h*)p, *(const v8h*)(p + 16));
          acc[nt] = wmma_f16(qf[ks], kf, acc[nt]);
        }
      }
      // online softmax per row (rows live in 16-lane halves; xor 1/2/4/8)
      _Float16* Pw = Psm + wid * 16 * 40;
#pragma unroll
      for (int r = 0; r < 8; ++r) {
        int qi = qRow0 + r + 8 * hi;
        int kj = kb * 32 + l16;
        float s0 = (kj      <= qi) ? acc[0][r] : -1e30f;
        float s1 = (kj + 16 <= qi) ? acc[1][r] : -1e30f;
        float mNew = fmaxf(rowM[r], rmax16(fmaxf(s0, s1)));
        float corr = __expf(rowM[r] - mNew);
        float p0 = __expf(s0 - mNew);
        float p1 = __expf(s1 - mNew);
        rowL[r] = rowL[r] * corr + rsum16(p0 + p1);
        rowM[r] = mNew;
#pragma unroll
        for (int nt = 0; nt < 4; ++nt) o[nt][r] *= corr;
        Pw[(r + 8 * hi) * 40 + l16]      = (_Float16)p0;
        Pw[(r + 8 * hi) * 40 + 16 + l16] = (_Float16)p1;
      }
      // re-layout P (C-layout -> A-fragment) through per-wave LDS scratch
      const _Float16* pp = Pw + l16 * 40;
      v16h pf = cat16(*(const v8h*)(pp + hi * 8), *(const v8h*)(pp + 16 + hi * 8));
      // O += P @ V
#pragma unroll
      for (int nt = 0; nt < 4; ++nt) {
        const _Float16* vp = Vts + (nt * 16 + l16) * 40 + hi * 8;
        v16h vf = cat16(*(const v8h*)vp, *(const v8h*)(vp + 16));
        o[nt] = wmma_f16(pf, vf, o[nt]);
      }
    }
    __syncthreads();
  }

  // epilogue: normalize and store merged-head layout [b, s, h*hd]
#pragma unroll
  for (int r = 0; r < 8; ++r) {
    float inv = 1.0f / rowL[r];
    int qi = qRow0 + r + 8 * hi;
    size_t base = (size_t)(b * NS + qi) * ND + h * NHD;
#pragma unroll
    for (int nt = 0; nt < 4; ++nt)
      O[base + nt * 16 + l16] = (_Float16)(o[nt][r] * inv);
  }
}

// ------------------------------- launcher ----------------------------------
extern "C" void kernel_launch(void* const* d_in, const int* in_sizes, int n_in,
                              void* d_out, int out_size, void* d_ws, size_t ws_size,
                              hipStream_t stream) {
  const float* hidden   = (const float*)d_in[0];   // [4,2048,1024]
  const float* c_attn_w = (const float*)d_in[1];   // [1024,3072]
  const float* c_attn_b = (const float*)d_in[2];   // [3072]
  const float* c_proj_w = (const float*)d_in[3];   // [1024,1024]
  const float* c_proj_b = (const float*)d_in[4];   // [1024]
  float* out = (float*)d_out;                      // [4,2048,1024] f32

  char* ws = (char*)d_ws;
  _Float16* Xh   = (_Float16*)(ws);                         // 16 MiB [8192,1024]
  _Float16* W1t  = (_Float16*)(ws + 16777216);              //  6 MiB [3072,1024]
  _Float16* W2t  = (_Float16*)(ws + 23068672);              //  2 MiB [1024,1024]
  _Float16* QKVh = (_Float16*)(ws + 25165824);              // 48 MiB [8192,3072]
  _Float16* AttO = (_Float16*)(ws + 75497472);              // 16 MiB [8192,1024]

  // 1) precision / layout prep
  cvt_f32_f16<<<(NTOK * ND / 4 + 255) / 256, 256, 0, stream>>>(hidden, Xh, NTOK * ND / 4);
  xpose_f32_f16<<<dim3(3 * ND / 32, ND / 32), 256, 0, stream>>>(c_attn_w, W1t, ND, 3 * ND);
  xpose_f32_f16<<<dim3(ND / 32, ND / 32), 256, 0, stream>>>(c_proj_w, W2t, ND, ND);

  // 2) fused QKV projection (+bias, Q pre-scaled by 1/8)
  gemm_wmma<1, 1><<<dim3(3 * ND / 128, NTOK / 128), 256, 0, stream>>>(
      Xh, W1t, c_attn_b, (void*)QKVh, NTOK, 3 * ND, ND);

  // 3) causal flash attention, all heads
  attn_wmma<<<dim3(NS / 128, NB * NH), 256, 0, stream>>>(QKVh, AttO);

  // 4) output projection (+bias) -> fp32
  gemm_wmma<0, 0><<<dim3(ND / 128, NTOK / 128), 256, 0, stream>>>(
      AttO, W2t, c_proj_b, (void*)out, NTOK, ND, ND);
}
